// GraphAttentionEmbedding_26379689132034
// MI455X (gfx1250) — matrix operand order repaired and checked by reading, not compiled, not run on hardware
//
#include <hip/hip_runtime.h>
#include <hip/hip_bf16.h>
#include <math.h>

#define N_NODES 50000
#define N_EDGES 800000
#define DIM 128
#define EDIM 96
#define TDIM 32
#define MSGD 64

typedef __attribute__((ext_vector_type(16))) __bf16 v16bf;
typedef __attribute__((ext_vector_type(8)))  float  v8f;
typedef __attribute__((ext_vector_type(4)))  unsigned uint4v;
typedef __attribute__((ext_vector_type(2)))  float  float2v;

// ---------------- helpers ----------------

__device__ inline unsigned short f32_to_bf16_bits(float f) {
  unsigned u = __builtin_bit_cast(unsigned, f);
  unsigned r = u + 0x7FFFu + ((u >> 16) & 1u);   // round-to-nearest-even
  return (unsigned short)(r >> 16);
}

// order-preserving float -> u32 map for atomic max
__device__ inline unsigned mapf(float f) {
  unsigned u = __builtin_bit_cast(unsigned, f);
  return (u & 0x80000000u) ? ~u : (u | 0x80000000u);
}
__device__ inline float unmapf(unsigned m) {
  unsigned u = (m & 0x80000000u) ? (m & 0x7FFFFFFFu) : ~m;
  return __builtin_bit_cast(float, u);
}

// Load one pre-packed fragment: 16 contiguous bf16 per lane -> two b128 loads.
// Packed layout: base[(frag*32 + lane)*16 + j] for j=0..15.
__device__ inline v16bf load_frag(const unsigned short* p, int frag, int lane) {
  union { uint4v u[2]; v16bf v; } f;
  const uint4v* q = (const uint4v*)(p + (((unsigned)(frag * 32 + lane)) << 4));
  f.u[0] = q[0];
  f.u[1] = q[1];
  return f.v;
}

__device__ inline v8f wmma_bf16(v16bf a, v16bf b, v8f c) {
  return __builtin_amdgcn_wmma_f32_16x16x32_bf16(false, a, false, b, (short)0, c, false, false);
}

// A-matrix element map (ISA 7.12.2, 16-bit A 16x32):
//   lane = m + 16*hf ; element j -> kk = (j>>3)*16 + hf*8 + (j&7)
// inverse: given kk: hf = (kk>>3)&1 ; j = ((kk>>4)<<3) | (kk&7)
// B-matrix element map (32x16): lane = n + 16*hf ; element j -> kk = (j>>3)*16 + hf*8 + (j&7), as for A

// ---------------- pack kernels ----------------

// pack f32 W[K x 128] (row-major) into B-fragment order; K = 32*KT, frag = kt*8 + ct
__global__ __launch_bounds__(256) void k_pack_B(const float* __restrict__ W,
                                                unsigned short* __restrict__ out, int KT) {
  int idx = blockIdx.x * blockDim.x + threadIdx.x;
  int total = KT * 8 * 32 * 16;
  if (idx >= total) return;
  int j = idx & 15, lane = (idx >> 4) & 31, frag = idx >> 9;
  int kt = frag >> 3, ct = frag & 7;
  int hf = lane >> 4, n = lane & 15;
  int k = kt * 32 + ((j >> 3) << 4) + (hf << 3) + (j & 7);
  out[idx] = f32_to_bf16_bits(W[(size_t)k * DIM + ct * 16 + n]);
}

// pack f32 x[N x 128] into A-fragment order per 16-row tile: frag = tile*4 + kt
__global__ __launch_bounds__(256) void k_pack_x(const float* __restrict__ x,
                                                unsigned short* __restrict__ out) {
  int idx = blockIdx.x * blockDim.x + threadIdx.x;
  if (idx >= N_NODES * DIM) return;
  int j = idx & 15, lane = (idx >> 4) & 31, frag = idx >> 9;
  int kt = frag & 3, tile = frag >> 2;
  int m = lane & 15, hf = lane >> 4;
  int kk = ((j >> 3) << 4) + (hf << 3) + (j & 7);
  out[idx] = f32_to_bf16_bits(x[(size_t)(tile * 16 + m) * DIM + kt * 32 + kk]);
}

// ---------------- init ----------------

__global__ void k_init(float* __restrict__ accum, float* __restrict__ denom,
                       unsigned* __restrict__ amax) {
  long long i = (long long)blockIdx.x * blockDim.x + threadIdx.x;
  if (i < (long long)N_NODES * DIM) accum[i] = 0.0f;
  if (i < 2LL * N_NODES) { denom[i] = 0.0f; amax[i] = 0u; }
}

// ---------------- node projections ----------------

__global__ __launch_bounds__(128) void k_node_proj(
    const unsigned short* __restrict__ xpack,
    const unsigned short* __restrict__ wqp, const unsigned short* __restrict__ wkp,
    const unsigned short* __restrict__ wvp,
    const float* __restrict__ b_q, const float* __restrict__ b_k,
    const float* __restrict__ b_v,
    float* __restrict__ qbuf, float* __restrict__ kbuf, float* __restrict__ vbuf) {
  int tid = threadIdx.x;
  int wave = tid >> 5, lane = tid & 31;
  int n = lane & 15, hf = lane >> 4;
  int tile = blockIdx.x, m0 = tile * 16;
  const unsigned short* ab = xpack + ((size_t)tile << 11);  // 4 frags * 512
  v16bf a0 = load_frag(ab, 0, lane);
  v16bf a1 = load_frag(ab, 1, lane);
  v16bf a2 = load_frag(ab, 2, lane);
  v16bf a3 = load_frag(ab, 3, lane);
  for (int item = wave; item < 24; item += 4) {
    int mat = item >> 3, ct = item & 7, n0 = ct * 16;
    const unsigned short* wp = (mat == 0) ? wqp : ((mat == 1) ? wkp : wvp);
    const float* bb = (mat == 0) ? b_q : ((mat == 1) ? b_k : b_v);
    float* out = (mat == 0) ? qbuf : ((mat == 1) ? kbuf : vbuf);
    // issue all B loads as one clause, then back-to-back WMMAs
    v16bf bb0 = load_frag(wp, 0 * 8 + ct, lane);
    v16bf bb1 = load_frag(wp, 1 * 8 + ct, lane);
    v16bf bb2 = load_frag(wp, 2 * 8 + ct, lane);
    v16bf bb3 = load_frag(wp, 3 * 8 + ct, lane);
    float bias = bb[n0 + n];
    v8f acc;
#pragma unroll
    for (int r = 0; r < 8; ++r) acc[r] = bias;
    acc = wmma_bf16(a0, bb0, acc);
    acc = wmma_bf16(a1, bb1, acc);
    acc = wmma_bf16(a2, bb2, acc);
    acc = wmma_bf16(a3, bb3, acc);
#pragma unroll
    for (int r = 0; r < 8; ++r)
      out[(size_t)(m0 + hf * 8 + r) * DIM + n0 + n] = acc[r];
  }
}

// ---------------- edge staging (64 edges / block, packed attr in LDS) ----------------

__device__ inline void stage_block(
    int tid, long long ebase0,
    const long long* __restrict__ ei, const float* __restrict__ lu,
    const float* __restrict__ tt, const float* __restrict__ msg,
    const float* __restrict__ W_time, const float* __restrict__ b_time,
    int* s_src, int* s_dst, float* s_relt, unsigned short* s_attr) {
  if (tid < 64) {
    long long e = ebase0 + tid;
    int s = (int)ei[e];
    s_src[tid] = s;
    s_relt[tid] = lu[s] - tt[e];
  } else {
    long long e = ebase0 + (tid - 64);
    s_dst[tid - 64] = (int)ei[(long long)N_EDGES + e];
  }
  // prefetch msg region of the block 8 ahead (lowers to global_prefetch_b8)
  {
    long long enext = ebase0 + 8LL * 64;
    if (enext + 64 <= N_EDGES)
      __builtin_prefetch(msg + enext * MSGD + (long long)tid * 32, 0, 0);
  }
  __syncthreads();
  // fill packed attr: 64 edges x 96 features, written in A-fragment order
  for (int i = tid; i < 64 * EDIM; i += 128) {
    int m64 = i / EDIM, c = i - m64 * EDIM;
    float val;
    if (c < TDIM) val = cosf(s_relt[m64] * W_time[c] + b_time[c]);
    else          val = msg[(size_t)(ebase0 + m64) * MSGD + (c - TDIM)];
    int tib = m64 >> 4, m = m64 & 15;
    int kt = c >> 5, kk = c & 31;
    int lanep = m | ((kk & 8) << 1);                 // m + 16*((kk>>3)&1)
    int jp = ((kk >> 4) << 3) | (kk & 7);
    s_attr[(((tib * 3 + kt) * 32 + lanep) << 4) + jp] = f32_to_bf16_bits(val);
  }
  __syncthreads();
}

// ---------------- edge pass 1: alpha + segment max ----------------

__global__ __launch_bounds__(128) void k_edge_alpha(
    const long long* __restrict__ ei, const float* __restrict__ lu,
    const float* __restrict__ tt, const float* __restrict__ msg,
    const float* __restrict__ W_time, const float* __restrict__ b_time,
    const unsigned short* __restrict__ wep, const float* __restrict__ b_e,
    const float* __restrict__ qbuf, const float* __restrict__ kbuf,
    float* __restrict__ alpha, unsigned* __restrict__ amax) {
  __shared__ __align__(32) unsigned short s_attr[4 * 3 * 32 * 16];  // 12 KB
  __shared__ int s_src[64], s_dst[64];
  __shared__ float s_relt[64];
  int tid = threadIdx.x;
  long long ebase0 = (long long)blockIdx.x * 64;
  stage_block(tid, ebase0, ei, lu, tt, msg, W_time, b_time,
              s_src, s_dst, s_relt, s_attr);

  int wave = tid >> 5, lane = tid & 31;
  int n = lane & 15, hf = lane >> 4;
  const unsigned short* ab = s_attr + wave * 3 * 512;
  const int* src = s_src + wave * 16;
  const int* dst = s_dst + wave * 16;
  long long ebase = ebase0 + wave * 16;

  v16bf a0 = load_frag(ab, 0, lane);
  v16bf a1 = load_frag(ab, 1, lane);
  v16bf a2 = load_frag(ab, 2, lane);

  float al0[8], al1[8];
#pragma unroll
  for (int r = 0; r < 8; ++r) { al0[r] = 0.0f; al1[r] = 0.0f; }

  // double-buffered B fragments across column tiles
  v16bf nb0 = load_frag(wep, 0 * 8 + 0, lane);
  v16bf nb1 = load_frag(wep, 1 * 8 + 0, lane);
  v16bf nb2 = load_frag(wep, 2 * 8 + 0, lane);
#pragma unroll
  for (int ct = 0; ct < 8; ++ct) {
    v16bf c0 = nb0, c1 = nb1, c2 = nb2;
    if (ct < 7) {
      nb0 = load_frag(wep, 0 * 8 + ct + 1, lane);
      nb1 = load_frag(wep, 1 * 8 + ct + 1, lane);
      nb2 = load_frag(wep, 2 * 8 + ct + 1, lane);
    }
    int n0 = ct * 16;
    float bias = b_e[n0 + n];
    v8f acc;
#pragma unroll
    for (int r = 0; r < 8; ++r) acc[r] = bias;
    acc = wmma_bf16(a0, c0, acc);
    acc = wmma_bf16(a1, c1, acc);
    acc = wmma_bf16(a2, c2, acc);
    int col = n0 + n;
#pragma unroll
    for (int r = 0; r < 8; ++r) {
      int m = r + hf * 8;
      int sm = src[m], dm = dst[m];
      float kv = kbuf[(size_t)sm * DIM + col] + acc[r];
      float qv = qbuf[(size_t)dm * DIM + col];
      if (ct < 4) al0[r] += qv * kv; else al1[r] += qv * kv;
    }
  }
  const float scale = 0.125f;  // 1/sqrt(64)
#pragma unroll
  for (int r = 0; r < 8; ++r) {
    float a0s = al0[r], a1s = al1[r];
#pragma unroll
    for (int msk = 1; msk < 16; msk <<= 1) {
      a0s += __shfl_xor(a0s, msk);
      a1s += __shfl_xor(a1s, msk);
    }
    if (n == 0) {
      int m = r + hf * 8;
      a0s *= scale; a1s *= scale;
      float2v av; av.x = a0s; av.y = a1s;
      ((float2v*)alpha)[ebase + m] = av;
      int dm = dst[m];
      atomicMax(&amax[dm * 2 + 0], mapf(a0s));
      atomicMax(&amax[dm * 2 + 1], mapf(a1s));
    }
  }
}

// ---------------- softmax denom (one edge / thread, both heads) ----------------

__global__ __launch_bounds__(256) void k_softmax_denom(
    const long long* __restrict__ ei, const float* __restrict__ alpha,
    const unsigned* __restrict__ amax, float* __restrict__ ex,
    float* __restrict__ denom) {
  long long e = (long long)blockIdx.x * blockDim.x + threadIdx.x;
  if (e >= (long long)N_EDGES) return;
  int dm = (int)ei[(long long)N_EDGES + e];
  float2v al = ((const float2v*)alpha)[e];
  float am0 = unmapf(amax[dm * 2 + 0]);
  float am1 = unmapf(amax[dm * 2 + 1]);
  float v0 = expf(al.x - am0);
  float v1 = expf(al.y - am1);
  float2v ev; ev.x = v0; ev.y = v1;
  ((float2v*)ex)[e] = ev;
  atomicAdd(&denom[dm * 2 + 0], v0);
  atomicAdd(&denom[dm * 2 + 1], v1);
}

// ---------------- edge pass 2: aggregate numerator ----------------

__global__ __launch_bounds__(128) void k_edge_aggregate(
    const long long* __restrict__ ei, const float* __restrict__ lu,
    const float* __restrict__ tt, const float* __restrict__ msg,
    const float* __restrict__ W_time, const float* __restrict__ b_time,
    const unsigned short* __restrict__ wep, const float* __restrict__ b_e,
    const float* __restrict__ vbuf, const float* __restrict__ ex,
    float* __restrict__ accum) {
  __shared__ __align__(32) unsigned short s_attr[4 * 3 * 32 * 16];
  __shared__ int s_src[64], s_dst[64];
  __shared__ float s_relt[64];
  int tid = threadIdx.x;
  long long ebase0 = (long long)blockIdx.x * 64;
  stage_block(tid, ebase0, ei, lu, tt, msg, W_time, b_time,
              s_src, s_dst, s_relt, s_attr);

  int wave = tid >> 5, lane = tid & 31;
  int n = lane & 15, hf = lane >> 4;
  const unsigned short* ab = s_attr + wave * 3 * 512;
  const int* src = s_src + wave * 16;
  const int* dst = s_dst + wave * 16;
  long long ebase = ebase0 + wave * 16;

  v16bf a0 = load_frag(ab, 0, lane);
  v16bf a1 = load_frag(ab, 1, lane);
  v16bf a2 = load_frag(ab, 2, lane);

  // hoist ex gathers (per edge, both heads), reused across 8 column tiles
  float ex0[8], ex1[8];
#pragma unroll
  for (int r = 0; r < 8; ++r) {
    float2v ev = ((const float2v*)ex)[ebase + r + hf * 8];
    ex0[r] = ev.x;
    ex1[r] = ev.y;
  }

  // double-buffered B fragments across column tiles
  v16bf nb0 = load_frag(wep, 0 * 8 + 0, lane);
  v16bf nb1 = load_frag(wep, 1 * 8 + 0, lane);
  v16bf nb2 = load_frag(wep, 2 * 8 + 0, lane);
#pragma unroll
  for (int ct = 0; ct < 8; ++ct) {
    v16bf c0 = nb0, c1 = nb1, c2 = nb2;
    if (ct < 7) {
      nb0 = load_frag(wep, 0 * 8 + ct + 1, lane);
      nb1 = load_frag(wep, 1 * 8 + ct + 1, lane);
      nb2 = load_frag(wep, 2 * 8 + ct + 1, lane);
    }
    int n0 = ct * 16;
    float bias = b_e[n0 + n];
    v8f acc;
#pragma unroll
    for (int r = 0; r < 8; ++r) acc[r] = bias;
    acc = wmma_bf16(a0, c0, acc);
    acc = wmma_bf16(a1, c1, acc);
    acc = wmma_bf16(a2, c2, acc);
    int col = n0 + n;
#pragma unroll
    for (int r = 0; r < 8; ++r) {
      int m = r + hf * 8;
      int sm = src[m], dm = dst[m];
      float w = (ct < 4) ? ex0[r] : ex1[r];
      float val = w * (vbuf[(size_t)sm * DIM + col] + acc[r]);
      atomicAdd(&accum[(size_t)dm * DIM + col], val);
    }
  }
}

// ---------------- final: skip GEMM + normalize ----------------

__global__ __launch_bounds__(128) void k_final(
    const unsigned short* __restrict__ xpack, const unsigned short* __restrict__ wskp,
    const float* __restrict__ b_skip, const float* __restrict__ accum,
    const float* __restrict__ denom, float* __restrict__ out) {
  int tid = threadIdx.x;
  int wave = tid >> 5, lane = tid & 31;
  int n = lane & 15, hf = lane >> 4;
  int tile = blockIdx.x, m0 = tile * 16;
  const unsigned short* ab = xpack + ((size_t)tile << 11);
  v16bf a0 = load_frag(ab, 0, lane);
  v16bf a1 = load_frag(ab, 1, lane);
  v16bf a2 = load_frag(ab, 2, lane);
  v16bf a3 = load_frag(ab, 3, lane);
  for (int ct = wave; ct < 8; ct += 4) {
    int n0 = ct * 16, head = ct >> 2;
    v16bf bb0 = load_frag(wskp, 0 * 8 + ct, lane);
    v16bf bb1 = load_frag(wskp, 1 * 8 + ct, lane);
    v16bf bb2 = load_frag(wskp, 2 * 8 + ct, lane);
    v16bf bb3 = load_frag(wskp, 3 * 8 + ct, lane);
    float bias = b_skip[n0 + n];
    v8f acc;
#pragma unroll
    for (int r = 0; r < 8; ++r) acc[r] = bias;
    acc = wmma_bf16(a0, bb0, acc);
    acc = wmma_bf16(a1, bb1, acc);
    acc = wmma_bf16(a2, bb2, acc);
    acc = wmma_bf16(a3, bb3, acc);
#pragma unroll
    for (int r = 0; r < 8; ++r) {
      int row = m0 + hf * 8 + r;
      float den = denom[row * 2 + head] + 1e-16f;
      out[(size_t)row * DIM + n0 + n] =
          acc[r] + accum[(size_t)row * DIM + n0 + n] / den;
    }
  }
}

// ---------------- launch ----------------

extern "C" void kernel_launch(void* const* d_in, const int* in_sizes, int n_in,
                              void* d_out, int out_size, void* d_ws, size_t ws_size,
                              hipStream_t stream) {
  const float*     x      = (const float*)d_in[0];
  const float*     lu     = (const float*)d_in[1];
  const float*     tt     = (const float*)d_in[2];
  const float*     msg    = (const float*)d_in[3];
  const long long* ei     = (const long long*)d_in[4];
  const float*     W_time = (const float*)d_in[5];
  const float*     b_time = (const float*)d_in[6];
  const float*     W_q    = (const float*)d_in[7];
  const float*     b_q    = (const float*)d_in[8];
  const float*     W_k    = (const float*)d_in[9];
  const float*     b_k    = (const float*)d_in[10];
  const float*     W_v    = (const float*)d_in[11];
  const float*     b_v    = (const float*)d_in[12];
  const float*     W_e    = (const float*)d_in[13];
  const float*     b_e    = (const float*)d_in[14];
  const float*     W_sk   = (const float*)d_in[15];
  const float*     b_sk   = (const float*)d_in[16];
  float* out = (float*)d_out;

  float* ws = (float*)d_ws;
  const size_t ND = (size_t)N_NODES * DIM;
  size_t off = 0;
  auto take = [&](size_t nfloats) {
    float* p = ws + off;
    off += (nfloats + 15) & ~(size_t)15;  // 64B alignment
    return p;
  };
  float*    qbuf  = take(ND);
  float*    kbuf  = take(ND);
  float*    vbuf  = take(ND);
  float*    accum = take(ND);
  float*    alpha = take(2 * (size_t)N_EDGES);
  float*    exb   = take(2 * (size_t)N_EDGES);
  unsigned* amax  = (unsigned*)take(2 * (size_t)N_NODES);
  float*    denom = take(2 * (size_t)N_NODES);
  unsigned short* xpack = (unsigned short*)take(ND / 2);   // ND bf16
  unsigned short* wqp   = (unsigned short*)take(8192);     // 16384 bf16
  unsigned short* wkp   = (unsigned short*)take(8192);
  unsigned short* wvp   = (unsigned short*)take(8192);
  unsigned short* wskp  = (unsigned short*)take(8192);
  unsigned short* wep   = (unsigned short*)take(6144);     // 12288 bf16

  // init accum/denom/amax
  {
    long long total = (long long)N_NODES * DIM;
    k_init<<<(int)((total + 255) / 256), 256, 0, stream>>>(accum, denom, amax);
  }
  // pack operands into WMMA fragment layout (bf16)
  k_pack_x<<<(N_NODES * DIM + 255) / 256, 256, 0, stream>>>(x, xpack);
  k_pack_B<<<(4 * 4096 + 255) / 256, 256, 0, stream>>>(W_q, wqp, 4);
  k_pack_B<<<(4 * 4096 + 255) / 256, 256, 0, stream>>>(W_k, wkp, 4);
  k_pack_B<<<(4 * 4096 + 255) / 256, 256, 0, stream>>>(W_v, wvp, 4);
  k_pack_B<<<(4 * 4096 + 255) / 256, 256, 0, stream>>>(W_sk, wskp, 4);
  k_pack_B<<<(3 * 4096 + 255) / 256, 256, 0, stream>>>(W_e, wep, 3);
  // node projections
  k_node_proj<<<N_NODES / 16, 128, 0, stream>>>(xpack, wqp, wkp, wvp,
                                                b_q, b_k, b_v, qbuf, kbuf, vbuf);
  // edge pass 1: alpha + segment max
  k_edge_alpha<<<N_EDGES / 64, 128, 0, stream>>>(ei, lu, tt, msg, W_time, b_time,
                                                 wep, b_e, qbuf, kbuf, alpha, amax);
  // softmax denom
  k_softmax_denom<<<(N_EDGES + 255) / 256, 256, 0, stream>>>(ei, alpha, amax,
                                                             exb, denom);
  // edge pass 2: aggregate numerator
  k_edge_aggregate<<<N_EDGES / 64, 128, 0, stream>>>(ei, lu, tt, msg, W_time, b_time,
                                                     wep, b_e, vbuf, exb, accum);
  // final: skip GEMM + normalize
  k_final<<<N_NODES / 16, 128, 0, stream>>>(xpack, wskp, b_sk, accum, denom, out);
}
